// AutoEncoder_13597866459299
// MI455X (gfx1250) — matrix operand order repaired and checked by reading, not compiled, not run on hardware
//
#include <hip/hip_runtime.h>
#include <hip/hip_bf16.h>
#include <math.h>

typedef __attribute__((ext_vector_type(16))) _Float16 v16h;
typedef __attribute__((ext_vector_type(8)))  float    v8f;

#define TM 128
#define TN 64
#define KT 32
#define NSLOT_A (TM * KT / 256)   // 16
#define NSLOT_B (TN * KT / 256)   // 8

// ---------------------------------------------------------------------------
// Universal implicit-GEMM conv1d / linear kernel, WMMA f32_16x16x32_f16.
//   out[b, co, l] = act( bias[co] + sum_{ci,k} w[co,ci,k] * in[b, ci, l+k-pad] )
// Macro-tile 128x64; 8 waves each compute a 32x32 region (4 WMMAs / K-step).
// Double-buffered LDS -> one barrier per K-step. KW templated so the only
// inner-loop integer division folds to a magic multiply. Staging is fully
// branchless: address-select + value-select, loads batched for MLP.
// ---------------------------------------------------------------------------
template <int KW>
__global__ __launch_bounds__(256) void conv1d_wmma(
    const float* __restrict__ in, long long sBi, long long sCi, long long sLi,
    const float* __restrict__ w,  const float* __restrict__ bias,
    float* __restrict__ out, long long sBo, long long sCo, long long sLo,
    int Bt, int Cin, int Cout, int L, int pad, int act, int accumulate)
{
    const int Ktot = Cin * KW;
    const int M    = Bt * L;

    __shared__ _Float16 ldsA[2][TM][KT];
    __shared__ _Float16 ldsB[2][TN][KT];

    const int tid  = threadIdx.x;
    const int bM   = blockIdx.x * TM;
    const int bN   = blockIdx.y * TN;
    const int wave = tid >> 5;
    const int lane = tid & 31;
    const int wm   = wave & 3;            // 4 waves along M (32 rows each)
    const int wn   = wave >> 2;           // 2 waves along N (32 cols each)

    // ---- hoisted per-slot row info for the A loader (loop-invariant) ----
    const int cK  = tid & 31;             // K-offset within tile (all slots)
    const int rA0 = tid >> 5;             // A rows: rA0 + 8*j
    const float* rowPtr[NSLOT_A];
    int          rowL[NSLOT_A];           // l; huge value marks invalid row
    #pragma unroll
    for (int j = 0; j < NSLOT_A; ++j) {
        int m = bM + rA0 + j * 8;
        if (m < M) {
            int b = m / L;                // once per kernel, not per K-step
            int l = m - b * L;
            rowPtr[j] = in + (long long)b * sBi + (long long)l * sLi;
            rowL[j]   = l;
        } else {
            rowPtr[j] = in;
            rowL[j]   = 0x40000000;       // pos check always fails
        }
    }
    // ---- hoisted weight-row pointers for the B loader ----
    const float* wRow[NSLOT_B];
    #pragma unroll
    for (int j = 0; j < NSLOT_B; ++j) {
        int co = bN + rA0 + j * 8;        // B cols: rA0 + 8*j
        wRow[j] = (co < Cout) ? (w + (long long)co * Ktot) : nullptr;
    }

    v8f acc[2][2];
    #pragma unroll
    for (int i = 0; i < 2; ++i)
        #pragma unroll
        for (int j = 0; j < 2; ++j)
            acc[i][j] = v8f{0.f,0.f,0.f,0.f,0.f,0.f,0.f,0.f};

    // Branchless staging: issue ALL loads first (address clamped via select so
    // every lane loads from a valid location), then mask, convert and store.
    auto stage = [&](int k0, int buf) {
        int  kk     = k0 + cK;
        int  ci     = kk / KW;            // constant divisor -> magic multiply
        int  dk     = kk - ci * KW;
        bool kvalid = kk < Ktot;
        long long aoff = (long long)ci * sCi + (long long)(dk - pad) * sLi;

        float va[NSLOT_A];
        bool  oka[NSLOT_A];
        #pragma unroll
        for (int j = 0; j < NSLOT_A; ++j) {
            int  pos = rowL[j] + dk - pad;
            oka[j]   = kvalid & (pos >= 0) & (pos < L);
            const float* p = oka[j] ? (rowPtr[j] + aoff) : in;
            va[j] = *p;                   // unconditional, pipelined load
        }
        float vb[NSLOT_B];
        bool  okb[NSLOT_B];
        #pragma unroll
        for (int j = 0; j < NSLOT_B; ++j) {
            okb[j] = kvalid & (wRow[j] != nullptr);
            const float* p = okb[j] ? (wRow[j] + kk) : w;
            vb[j] = *p;
        }
        #pragma unroll
        for (int j = 0; j < NSLOT_A; ++j)
            ldsA[buf][rA0 + j * 8][cK] = (_Float16)(oka[j] ? va[j] : 0.f);
        #pragma unroll
        for (int j = 0; j < NSLOT_B; ++j)
            ldsB[buf][rA0 + j * 8][cK] = (_Float16)(okb[j] ? vb[j] : 0.f);
    };

    const int nK = (Ktot + KT - 1) / KT;
    stage(0, 0);
    int cur = 0;
    const int hi = (lane >= 16) ? 1 : 0;
    const int mr = wm * 32 + (lane & 15);
    const int nc = wn * 32 + (lane & 15);

    for (int t = 0; t < nK; ++t) {
        __syncthreads();
        if (t + 1 < nK) {
            __builtin_prefetch(&w[(size_t)bN * Ktot + (t + 1) * KT], 0, 1);
            stage((t + 1) * KT, cur ^ 1);
        }
        union { v16h v; unsigned u[8]; } a0, a1, b0, b1;
        #pragma unroll
        for (int i = 0; i < 8; ++i) {
            int ka = ((i < 4) ? (2 * i) : (16 + 2 * (i - 4))) + hi * 8;
            a0.u[i] = *(const unsigned*)&ldsA[cur][mr][ka];
            a1.u[i] = *(const unsigned*)&ldsA[cur][mr + 16][ka];
            int kb = ((i < 4) ? (2 * i) : (16 + 2 * (i - 4))) + hi * 8;
            b0.u[i] = *(const unsigned*)&ldsB[cur][nc][kb];
            b1.u[i] = *(const unsigned*)&ldsB[cur][nc + 16][kb];
        }
        acc[0][0] = __builtin_amdgcn_wmma_f32_16x16x32_f16(false, a0.v, false, b0.v, (short)0, acc[0][0], false, false);
        acc[0][1] = __builtin_amdgcn_wmma_f32_16x16x32_f16(false, a0.v, false, b1.v, (short)0, acc[0][1], false, false);
        acc[1][0] = __builtin_amdgcn_wmma_f32_16x16x32_f16(false, a1.v, false, b0.v, (short)0, acc[1][0], false, false);
        acc[1][1] = __builtin_amdgcn_wmma_f32_16x16x32_f16(false, a1.v, false, b1.v, (short)0, acc[1][1], false, false);
        cur ^= 1;
    }

    // ---- epilogue: bias + optional accumulate + activation, strided store ----
    #pragma unroll
    for (int mi = 0; mi < 2; ++mi) {
        #pragma unroll
        for (int ni = 0; ni < 2; ++ni) {
            int n = bN + wn * 32 + ni * 16 + (lane & 15);
            if (n < Cout) {
                float bv = bias ? bias[n] : 0.f;
                #pragma unroll
                for (int r = 0; r < 8; ++r) {
                    int m = bM + wm * 32 + mi * 16 + r + hi * 8;
                    if (m < M) {
                        int b = m / L;
                        int l = m - b * L;
                        long long o = (long long)b * sBo + (long long)n * sCo +
                                      (long long)l * sLo;
                        float v = acc[mi][ni][r] + bv;
                        if (accumulate) v += out[o];
                        if (act == 1)      v = fmaxf(v, 0.f);
                        else if (act == 2) v = v / (1.f + __expf(-v));
                        out[o] = v;
                    }
                }
            }
        }
    }
}

// ---------------------------------------------------------------------------
// project_in stage 1: conv1d(3->64, k7, pad3, L=20) + ReLU + MaxPool1d(4,4)
// ---------------------------------------------------------------------------
__global__ void conv1_relu_pool(const float* __restrict__ edge,
                                const float* __restrict__ w,   // [64,3,7]
                                const float* __restrict__ b,
                                float* __restrict__ out, int BN)
{
    int idx = blockIdx.x * 256 + threadIdx.x;
    if (idx >= BN * 64) return;
    int bn = idx >> 6, co = idx & 63;
    const float* e = edge + (long long)bn * 60;
    float wv[21];
    #pragma unroll
    for (int i = 0; i < 21; ++i) wv[i] = w[co * 21 + i];
    float bias = b[co];
    float o20[20];
    #pragma unroll
    for (int p = 0; p < 20; ++p) {
        float s = bias;
        #pragma unroll
        for (int ci = 0; ci < 3; ++ci)
            #pragma unroll
            for (int k = 0; k < 7; ++k) {
                int pos = p + k - 3;
                if (pos >= 0 && pos < 20) s += wv[ci * 7 + k] * e[pos * 3 + ci];
            }
        o20[p] = fmaxf(s, 0.f);
    }
    float* o = out + (long long)idx * 5;
    #pragma unroll
    for (int q = 0; q < 5; ++q)
        o[q] = fmaxf(fmaxf(o20[4*q], o20[4*q+1]), fmaxf(o20[4*q+2], o20[4*q+3]));
}

// mean over last dim of compact [Bt, C, L]  ->  out[Bt*C]
__global__ void mean_l(const float* __restrict__ in, float* __restrict__ out,
                       int BtC, int L)
{
    int idx = blockIdx.x * 256 + threadIdx.x;
    if (idx >= BtC) return;
    const float* p = in + (long long)idx * L;
    float s = 0.f;
    for (int l = 0; l < L; ++l) s += p[l];
    out[idx] = s / (float)L;
}

// ---------------- SAGE aggregation (segment mean via atomics) --------------
__global__ void scatter_cnt(const int* __restrict__ adj, float* __restrict__ cnt,
                            int B, int E, int N)
{
    int e = blockIdx.x * 256 + threadIdx.x;
    if (e >= B * E) return;
    int b = e / E;
    int dst = adj[(long long)e * 2 + 1] + b * N;
    atomicAdd(&cnt[dst], 1.f);
}

// 4 channels per thread: one adjacency lookup feeds a float4 + 4 atomics
__global__ void scatter_add4(const int* __restrict__ adj,
                             const float* __restrict__ h, float* __restrict__ agg,
                             int B, int E, int N, int C)
{
    int C4 = C >> 2;
    long long idx = (long long)blockIdx.x * 256 + threadIdx.x;
    long long tot = (long long)B * E * C4;
    if (idx >= tot) return;
    int e  = (int)(idx / C4);
    int c  = (int)(idx - (long long)e * C4) << 2;
    int b  = e / E;
    int src = adj[(long long)e * 2 + 0] + b * N;
    int dst = adj[(long long)e * 2 + 1] + b * N;
    const float4 v = *(const float4*)&h[(long long)src * C + c];
    float* a = &agg[(long long)dst * C + c];
    atomicAdd(a + 0, v.x);
    atomicAdd(a + 1, v.y);
    atomicAdd(a + 2, v.z);
    atomicAdd(a + 3, v.w);
}

__global__ void div_cnt(float* __restrict__ agg, const float* __restrict__ cnt,
                        int M, int C)
{
    long long idx = (long long)blockIdx.x * 256 + threadIdx.x;
    if (idx >= (long long)M * C) return;
    int m = (int)(idx / C);
    agg[idx] /= fmaxf(cnt[m], 1.f);
}

// L2-normalize rows of node-major [M, C]
__global__ void row_normalize(float* __restrict__ x, int M, int C, float eps)
{
    int m = blockIdx.x * 256 + threadIdx.x;
    if (m >= M) return;
    float* r = x + (long long)m * C;
    float s = 0.f;
    for (int c = 0; c < C; ++c) { float v = r[c]; s += v * v; }
    float sc = 1.f / fmaxf(sqrtf(s), eps);
    for (int c = 0; c < C; ++c) r[c] *= sc;
}

// h = LayerNorm(SiLU(h)) over rows of width C (C==64)
__global__ void silu_ln_rows(float* __restrict__ x, const float* __restrict__ w,
                             const float* __restrict__ b, int M, int C)
{
    int m = blockIdx.x * 256 + threadIdx.x;
    if (m >= M) return;
    float* r = x + (long long)m * C;
    float mean = 0.f;
    for (int c = 0; c < C; ++c) { float v = r[c]; v = v / (1.f + __expf(-v)); mean += v; }
    mean /= (float)C;
    float var = 0.f;
    for (int c = 0; c < C; ++c) { float v = r[c]; v = v / (1.f + __expf(-v)); float d = v - mean; var += d * d; }
    var /= (float)C;
    float rs = rsqrtf(var + 1e-5f);
    for (int c = 0; c < C; ++c) {
        float v = r[c]; v = v / (1.f + __expf(-v));
        r[c] = (v - mean) * rs * w[c] + b[c];
    }
}

// LayerNorm over the channel dim of [B, C, L]
__global__ void ln_channels(float* __restrict__ x, const float* __restrict__ w,
                            const float* __restrict__ b, int B, int C, int L)
{
    int idx = blockIdx.x * 256 + threadIdx.x;
    if (idx >= B * L) return;
    int bb = idx / L, l = idx - bb * L;
    float* p = x + (long long)bb * C * L + l;
    float mean = 0.f;
    for (int c = 0; c < C; ++c) mean += p[(long long)c * L];
    mean /= (float)C;
    float var = 0.f;
    for (int c = 0; c < C; ++c) { float d = p[(long long)c * L] - mean; var += d * d; }
    var /= (float)C;
    float rs = rsqrtf(var + 1e-5f);
    for (int c = 0; c < C; ++c)
        p[(long long)c * L] = (p[(long long)c * L] - mean) * rs * w[c] + b[c];
}

// PixelNorm(dim=1) then SiLU, in place on [B, C, L]
__global__ void pixelnorm_silu(float* __restrict__ x, int B, int C, int L)
{
    int idx = blockIdx.x * 256 + threadIdx.x;
    if (idx >= B * L) return;
    int bb = idx / L, l = idx - bb * L;
    float* p = x + (long long)bb * C * L + l;
    float s = 0.f;
    for (int c = 0; c < C; ++c) { float v = p[(long long)c * L]; s += v * v; }
    float sc = sqrtf((float)C) / fmaxf(sqrtf(s), 1e-4f);
    for (int c = 0; c < C; ++c) {
        float v = p[(long long)c * L] * sc;
        p[(long long)c * L] = v / (1.f + __expf(-v));
    }
}

// Squeeze-excite per batch row
__global__ void se_linear(const float* __restrict__ avg,
                          const float* __restrict__ s1w, const float* __restrict__ s1b,
                          const float* __restrict__ s2w, const float* __restrict__ s2b,
                          float* __restrict__ g, int d, int inner)
{
    __shared__ float hin[384];
    __shared__ float hmid[96];
    int b = blockIdx.x;
    for (int i = threadIdx.x; i < d; i += blockDim.x) hin[i] = avg[(long long)b * d + i];
    __syncthreads();
    for (int i = threadIdx.x; i < inner; i += blockDim.x) {
        float s = s1b[i];
        for (int k = 0; k < d; ++k) s += s1w[(long long)i * d + k] * hin[k];
        hmid[i] = s / (1.f + __expf(-s));
    }
    __syncthreads();
    for (int i = threadIdx.x; i < d; i += blockDim.x) {
        float s = s2b[i];
        for (int k = 0; k < inner; ++k) s += s2w[(long long)i * inner + k] * hmid[k];
        g[(long long)b * d + i] = 1.f / (1.f + __expf(-s));
    }
}

// out = h * g[b,c] + res  over [B, C, L]  (in-place safe when out==res)
__global__ void gate_res(const float* __restrict__ h, const float* __restrict__ g,
                         const float* __restrict__ res, float* __restrict__ out,
                         int B, int C, int L)
{
    long long idx = (long long)blockIdx.x * 256 + threadIdx.x;
    long long tot = (long long)B * C * L;
    if (idx >= tot) return;
    long long cl = idx % ((long long)C * L);
    int bb = (int)(idx / ((long long)C * L));
    int c  = (int)(cl / L);
    out[idx] = h[idx] * g[(long long)bb * C + c] + res[idx];
}

// ===========================================================================
extern "C" void kernel_launch(void* const* d_in, const int* in_sizes, int n_in,
                              void* d_out, int out_size, void* d_ws, size_t ws_size,
                              hipStream_t stream)
{
    (void)in_sizes; (void)n_in; (void)out_size; (void)ws_size;
    const int B = 64, N = 1024, E = 8192, BN = B * N;
    const long long M0 = BN;

    const float* edge = (const float*)d_in[0];
    const int*   adj  = (const int*)d_in[1];
    auto fp = [&](int i) { return (const float*)d_in[i]; };

    // params flattened alphabetically (tree-flatten order)
    const float* coor_b    = fp(2);
    const float* coor_w    = fp(3);
    const float* dec0_b    = fp(4);
    const float* dec0_ln_b = fp(5);
    const float* dec0_ln_w = fp(6);
    const float* dec0_w    = fp(7);
    const float* enc_ln_b  = fp(8);
    const float* enc_ln_w  = fp(9);
    const float* pi_c1_b   = fp(10);
    const float* pi_c1_w   = fp(11);
    const float* pi_c2_b   = fp(12);
    const float* pi_c2_w   = fp(13);
    const float* pi_l_b    = fp(14);
    const float* pi_l_w    = fp(15);

    struct ResW { const float *c1w,*c1b,*c2w,*c2b,*rcw,*rcb,*s1w,*s1b,*s2w,*s2b;
                  int din, dout, inner, has_rc; };
    const int dec_dims[17] = {128,128,128,128,192,192,192,192,256,256,256,256,256,256,384,384,384};
    ResW rb[16];
    int idx = 16, curr = dec_dims[0];
    for (int i = 0; i < 16; ++i) {
        int d = dec_dims[i + 1];
        rb[i].c1b = fp(idx++); rb[i].c1w = fp(idx++);
        rb[i].c2b = fp(idx++); rb[i].c2w = fp(idx++);
        rb[i].has_rc = (curr != d);
        rb[i].rcb = nullptr; rb[i].rcw = nullptr;
        if (rb[i].has_rc) { rb[i].rcb = fp(idx++); rb[i].rcw = fp(idx++); }
        rb[i].s1b = fp(idx++); rb[i].s1w = fp(idx++);
        rb[i].s2b = fp(idx++); rb[i].s2w = fp(idx++);
        rb[i].din = curr; rb[i].dout = d;
        rb[i].inner = (d / 4 > 16) ? d / 4 : 16;
        curr = d;
    }
    struct SageW { const float *lb,*lw,*pb,*pw,*rw; int ci, co; };
    SageW sg[4];
    const int sdims[4][2] = {{196,64},{64,128},{128,256},{256,256}};
    for (int i = 0; i < 4; ++i) {
        sg[i].lb = fp(idx++); sg[i].lw = fp(idx++);
        sg[i].pb = fp(idx++); sg[i].pw = fp(idx++);
        sg[i].rw = fp(idx++);
        sg[i].ci = sdims[i][0]; sg[i].co = sdims[i][1];
    }

    // ---- workspace arena (floats) ----
    const long long SLAB = (long long)64 * 384 * 1024;
    float* ws  = (float*)d_ws;
    float* hA  = ws;                       // M0*256
    float* hB  = hA  + M0 * 256;
    float* hC  = hB  + M0 * 256;
    float* agg = hC  + M0 * 256;
    float* cnt = agg + M0 * 256;           // M0
    float* avg = cnt + M0;                 // 64*384
    float* gbf = avg + (long long)64 * 384;
    float* xd0 = gbf + (long long)64 * 384;
    float* xd1 = xd0 + SLAB;
    float* xd2 = xd1 + SLAB;
    float* xd3 = xd2 + SLAB;
    // encoder-phase overlays inside decoder slabs
    float* t1  = xd0;                      // [BN,64,5]
    float* t2  = xd1;                      // [BN,128,5] (spans into xd2)
    float* t3  = xd3;                      // [BN,128]
    float* emb = xd3 + (long long)BN * 128;// [BN,196]

    auto conv = [&](const float* in, long long sBi, long long sCi, long long sLi,
                    const float* w, const float* bias, float* out,
                    long long sBo, long long sCo, long long sLo,
                    int Bt, int Cin, int Cout, int L, int Kw, int pad,
                    int act, int acc) {
        int M = Bt * L;
        dim3 grid((M + TM - 1) / TM, (Cout + TN - 1) / TN);
        if (Kw == 1)
            conv1d_wmma<1><<<grid, dim3(256), 0, stream>>>(
                in, sBi, sCi, sLi, w, bias, out, sBo, sCo, sLo,
                Bt, Cin, Cout, L, pad, act, acc);
        else if (Kw == 3)
            conv1d_wmma<3><<<grid, dim3(256), 0, stream>>>(
                in, sBi, sCi, sLi, w, bias, out, sBo, sCo, sLo,
                Bt, Cin, Cout, L, pad, act, acc);
        else
            conv1d_wmma<7><<<grid, dim3(256), 0, stream>>>(
                in, sBi, sCi, sLi, w, bias, out, sBo, sCo, sLo,
                Bt, Cin, Cout, L, pad, act, acc);
    };

    // ======================= project_in =======================
    conv1_relu_pool<<<(BN * 64 + 255) / 256, 256, 0, stream>>>(
        edge, pi_c1_w, pi_c1_b, t1, BN);
    conv(t1, 320, 5, 1, pi_c2_w, pi_c2_b, t2, 640, 5, 1,
         BN, 64, 128, 5, 3, 1, /*relu*/1, 0);
    mean_l<<<(BN * 128 + 255) / 256, 256, 0, stream>>>(t2, t3, BN * 128, 5);
    conv(t3, 0, 1, 128, pi_l_w, pi_l_b, emb, 0, 1, 196,
         1, 128, 196, BN, 1, 0, 0, 0);

    // ======================= SAGE layers =======================
    auto sage = [&](const float* x, float* out, const SageW& s) {
        int ci = s.ci, co = s.co;
        conv(x, 0, 1, ci, s.pw, s.pb, hA, 0, 1, ci,
             1, ci, ci, BN, 1, 0, /*relu*/1, 0);
        hipMemsetAsync(agg, 0, (size_t)M0 * ci * sizeof(float), stream);
        hipMemsetAsync(cnt, 0, (size_t)M0 * sizeof(float), stream);
        long long tot = (long long)B * E * (ci >> 2);
        scatter_add4<<<(unsigned)((tot + 255) / 256), 256, 0, stream>>>(
            adj, hA, agg, B, E, N, ci);
        scatter_cnt<<<(B * E + 255) / 256, 256, 0, stream>>>(adj, cnt, B, E, N);
        div_cnt<<<(unsigned)((M0 * ci + 255) / 256), 256, 0, stream>>>(
            agg, cnt, BN, ci);
        conv(agg, 0, 1, ci, s.lw, s.lb, out, 0, 1, co,
             1, ci, co, BN, 1, 0, 0, 0);
        conv(x, 0, 1, ci, s.rw, nullptr, out, 0, 1, co,
             1, ci, co, BN, 1, 0, 0, /*accumulate*/1);
        row_normalize<<<(BN + 255) / 256, 256, 0, stream>>>(out, BN, co, 1e-12f);
    };

    sage(emb, hB, sg[0]);
    silu_ln_rows<<<(BN + 255) / 256, 256, 0, stream>>>(hB, enc_ln_w, enc_ln_b, BN, 64);
    sage(hB, hC, sg[1]);
    sage(hC, hB, sg[2]);
    sage(hB, hC, sg[3]);   // hC: node-major [BN, 256]

    // ======================= decoder =======================
    conv(hC, (long long)1024 * 256, 1, 256, dec0_w, dec0_b,
         xd0, (long long)128 * 1024, 1024, 1,
         B, 256, 128, 1024, 7, 3, /*silu*/2, 0);
    ln_channels<<<(B * 1024 + 255) / 256, 256, 0, stream>>>(
        xd0, dec0_ln_w, dec0_ln_b, B, 128, 1024);

    for (int i = 0; i < 16; ++i) {
        const ResW& r = rb[i];
        int di = r.din, d = r.dout;
        if (r.has_rc)
            conv(xd0, (long long)di * 1024, 1024, 1, r.rcw, r.rcb,
                 xd3, (long long)d * 1024, 1024, 1,
                 B, di, d, 1024, 1, 0, 0, 0);
        conv(xd0, (long long)di * 1024, 1024, 1, r.c1w, r.c1b,
             xd1, (long long)d * 1024, 1024, 1,
             B, di, d, 1024, 3, 1, 0, 0);
        pixelnorm_silu<<<(B * 1024 + 255) / 256, 256, 0, stream>>>(xd1, B, d, 1024);
        conv(xd1, (long long)d * 1024, 1024, 1, r.c2w, r.c2b,
             xd2, (long long)d * 1024, 1024, 1,
             B, d, d, 1024, 3, 1, 0, 0);
        pixelnorm_silu<<<(B * 1024 + 255) / 256, 256, 0, stream>>>(xd2, B, d, 1024);
        mean_l<<<(B * d + 255) / 256, 256, 0, stream>>>(xd2, avg, B * d, 1024);
        se_linear<<<B, 256, 0, stream>>>(avg, r.s1w, r.s1b, r.s2w, r.s2b,
                                         gbf, d, r.inner);
        const float* res = r.has_rc ? xd3 : xd0;
        long long tot = (long long)B * d * 1024;
        gate_res<<<(unsigned)((tot + 255) / 256), 256, 0, stream>>>(
            xd2, gbf, res, xd0, B, d, 1024);
    }

    // final 384 -> 60 projection, writing [B, N, 60] directly to d_out
    conv(xd0, (long long)384 * 1024, 1024, 1, coor_w, coor_b,
         (float*)d_out, (long long)1024 * 60, 1, 60,
         B, 384, 60, 1024, 1, 0, 0, 0);
}
